// BoxMultiHeadedAttention_46213848106012
// MI455X (gfx1250) — matrix operand order repaired and checked by reading, not compiled, not run on hardware
//
#include <hip/hip_runtime.h>
#include <math.h>

// ---------------------------------------------------------------------------
// BoxMultiHeadedAttention for gfx1250 (MI455X).
// Projection GEMMs: f16 operands (pre-converted, weights pre-transposed to
// k-major), v_wmma_f32_16x16x32_f16 with f32 accumulate, BK=64, and
// GLOBAL_LOAD_ASYNC_TO_LDS_B128 tile staging when the builtin is available.
// ---------------------------------------------------------------------------

typedef _Float16 v4h  __attribute__((ext_vector_type(4)));
typedef _Float16 v8h  __attribute__((ext_vector_type(8)));
typedef _Float16 v16h __attribute__((ext_vector_type(16)));
typedef float    v8f  __attribute__((ext_vector_type(8)));
typedef int      v4i  __attribute__((ext_vector_type(4)));

typedef __attribute__((address_space(3))) v4i* lds_v4i;
typedef __attribute__((address_space(1))) v4i* glb_v4i;

#ifndef __has_builtin
#define __has_builtin(x) 0
#endif
#if __has_builtin(__builtin_amdgcn_global_load_async_to_lds_b128)
#define HAVE_ASYNC_LDS 1
#else
#define HAVE_ASYNC_LDS 0
#endif

static __device__ __forceinline__ void wait_async0() {
#if __has_builtin(__builtin_amdgcn_s_wait_asynccnt)
    __builtin_amdgcn_s_wait_asynccnt(0);
#else
    asm volatile("s_wait_asynccnt 0x0" ::: "memory");
#endif
}

#define GB   1024          // batch
#define GN   36            // boxes
#define GH   16            // heads
#define GDK  64            // head dim
#define GD   1024          // model dim
#define GM   (GB * GN)     // 36864 rows

// LDS tile row stride in halfs: 144 bytes, 16B aligned; r*36 mod 64 banks is
// a full 16-lane rotation -> conflict-free b128 fragment loads.
#define LDK 72

// ===========================================================================
// f32 -> f16 streaming convert (4 elements / thread, b128 in, b64 out).
// ===========================================================================
__global__ __launch_bounds__(256)
void cvt_f16(const float* __restrict__ src, _Float16* __restrict__ dst, int n4) {
    int i = blockIdx.x * 256 + threadIdx.x;
    if (i < n4) {
        const float4 f = ((const float4*)src)[i];
        v4h h;
        h[0] = (_Float16)f.x; h[1] = (_Float16)f.y;
        h[2] = (_Float16)f.z; h[3] = (_Float16)f.w;
        *(v4h*)&dst[(size_t)i * 4] = h;
    }
}

// ===========================================================================
// W [K,N] f32 -> Wt [N,K] f16 (transpose via LDS 32x32 tile).
// ===========================================================================
__global__ __launch_bounds__(256)
void cvt_t_f16(const float* __restrict__ W, _Float16* __restrict__ Wt,
               int K, int N) {
    __shared__ float tile[32][33];
    const int n0 = blockIdx.x * 32;
    const int k0 = blockIdx.y * 32;
    const int t  = threadIdx.x;
#pragma unroll
    for (int i = 0; i < 4; ++i) {
        int e = t + i * 256, r = e >> 5, c = e & 31;
        tile[r][c] = W[(size_t)(k0 + r) * N + n0 + c];
    }
    __syncthreads();
#pragma unroll
    for (int i = 0; i < 4; ++i) {
        int e = t + i * 256, r = e >> 5, c = e & 31;
        Wt[(size_t)(n0 + r) * K + k0 + c] = (_Float16)tile[c][r];
    }
}

// ===========================================================================
// GEMM: Y[M,N] = A[M,K] @ Bt[N,K]^T + bias  (f16 operands, f32 accumulate)
// Block tile 128x128, BK=64. 8 waves, each wave computes 32x64 (2x4 WMMA
// accumulators, 2 k-steps per stage -> 16 WMMAs per LDS stage).
// ===========================================================================
__global__ __launch_bounds__(256)
void gemm_f16(const _Float16* __restrict__ A, const _Float16* __restrict__ Bt,
              const float* __restrict__ bias, float* __restrict__ Y,
              int M, int N, int K) {
    __shared__ __align__(16) _Float16 As[128 * LDK];   // [m][k]
    __shared__ __align__(16) _Float16 Bs[128 * LDK];   // [n][k]

    const int t    = threadIdx.x;
    const int wave = t >> 5;
    const int lane = t & 31;
    const int wm   = wave >> 1;          // 0..3 (32-row strip)
    const int wn   = wave & 1;           // 0..1 (64-col strip)
    const int bm   = blockIdx.x * 128;
    const int bn   = blockIdx.y * 128;
    const int r    = lane & 15;
    const int hi   = lane >> 4;

    // tile-staging coordinates: 1024 16B chunks per tile, 4 per thread
    const int srow = t >> 3;             // 0..31 (+32 per i)
    const int sc   = (t & 7) * 8;        // half offset within row

    v8f acc[2][4] = {};

    for (int k0 = 0; k0 < K; k0 += 64) {
#pragma unroll
        for (int i = 0; i < 4; ++i) {
            const int row = srow + i * 32;
            const _Float16* ga = A  + (size_t)(bm + row) * K + k0 + sc;
            const _Float16* gb = Bt + (size_t)(bn + row) * K + k0 + sc;
            _Float16* la = &As[row * LDK + sc];
            _Float16* lb = &Bs[row * LDK + sc];
#if HAVE_ASYNC_LDS
            __builtin_amdgcn_global_load_async_to_lds_b128(
                (glb_v4i)(void*)ga, (lds_v4i)la, 0, 0);
            __builtin_amdgcn_global_load_async_to_lds_b128(
                (glb_v4i)(void*)gb, (lds_v4i)lb, 0, 0);
#else
            *(v4i*)la = *(const v4i*)ga;
            *(v4i*)lb = *(const v4i*)gb;
#endif
        }
        if (k0 + 64 < K) {   // prefetch next k-tile into caches
            __builtin_prefetch(A  + (size_t)(bm + srow) * K + k0 + 64 + sc, 0, 1);
            __builtin_prefetch(Bt + (size_t)(bn + srow) * K + k0 + 64 + sc, 0, 1);
        }
#if HAVE_ASYNC_LDS
        wait_async0();
#endif
        __syncthreads();

#pragma unroll
        for (int kk = 0; kk < 64; kk += 32) {
            // A 16x32 f16: lane r = row M; halfs 0..7  = K kk+8*hi..,
            //                              halfs 8..15 = K kk+16+8*hi..
            v16h afrag[2];
#pragma unroll
            for (int i = 0; i < 2; ++i) {
                int row = wm * 32 + i * 16 + r;
                union { v16h v; v8h h[2]; } u;
                u.h[0] = *(const v8h*)&As[row * LDK + kk + 8 * hi];
                u.h[1] = *(const v8h*)&As[row * LDK + kk + 16 + 8 * hi];
                afrag[i] = u.v;
            }
            // B 32x16 f16: lane r = col N; halfs 0..7  = K kk+8*hi..,
            //                              halfs 8..15 = K kk+16+8*hi..
            v16h bfrag[4];
#pragma unroll
            for (int j = 0; j < 4; ++j) {
                int col = wn * 64 + j * 16 + r;
                union { v16h v; v8h h[2]; } u;
                u.h[0] = *(const v8h*)&Bs[col * LDK + kk + 8 * hi];
                u.h[1] = *(const v8h*)&Bs[col * LDK + kk + 16 + 8 * hi];
                bfrag[j] = u.v;
            }
#pragma unroll
            for (int i = 0; i < 2; ++i)
#pragma unroll
                for (int j = 0; j < 4; ++j)
                    acc[i][j] = __builtin_amdgcn_wmma_f32_16x16x32_f16(
                        false, afrag[i], false, bfrag[j],
                        (short)0, acc[i][j], false, false);
        }
        __syncthreads();
    }

    // C layout: lane r = N col, VGPR v -> M = v + 8*hi
#pragma unroll
    for (int i = 0; i < 2; ++i)
#pragma unroll
        for (int j = 0; j < 4; ++j) {
            int col = bn + wn * 64 + j * 16 + r;
            float bv = bias[col];
#pragma unroll
            for (int v = 0; v < 8; ++v) {
                int row = bm + wm * 32 + i * 16 + v + 8 * hi;
                Y[(size_t)row * N + col] = acc[i][j][v] + bv;
            }
        }
}

// ===========================================================================
// Geometry kernel: one workgroup per batch b.
// logw[b,h,m] = log(max(1e-6, sum_n relu(emb(b,n,m).Wg[h]+bg[h]) *
//                                (qh[b,h,n,:].Wa[:,m] + ba[m])))
// ===========================================================================
__global__ __launch_bounds__(256)
void geom_logw(const float* __restrict__ box, const float* __restrict__ Qp,
               const float* __restrict__ Wg, const float* __restrict__ bg,
               const float* __restrict__ Wa, const float* __restrict__ ba,
               float* __restrict__ logw) {
    __shared__ float cx[GN], cy[GN], ww[GN], hh[GN];
    __shared__ float WgS[GH * 64];
    __shared__ float bgS[GH];
    __shared__ float WaS[64 * GN];
    __shared__ float baS[GN];
    __shared__ float g[8 * GN * GN];
    __shared__ float wgacc[GH * GN];

    const int b = blockIdx.x;
    const int t = threadIdx.x;

    if (t < GN) {
        const float* bx = box + ((size_t)b * GN + t) * 4;
        float x0 = bx[0], y0 = bx[1], x1 = bx[2], y1 = bx[3];
        cx[t] = (x0 + x1) * 0.5f;
        cy[t] = (y0 + y1) * 0.5f;
        ww[t] = x1 - x0 + 1.0f;
        hh[t] = y1 - y0 + 1.0f;
    }
    for (int e = t; e < GH * 64; e += 256) WgS[e] = Wg[e];
    if (t < GH) bgS[t] = bg[t];
    for (int e = t; e < 64 * GN; e += 256) WaS[e] = Wa[e];
    if (t < GN) baS[t] = ba[t];
    __syncthreads();

    const float* qb = Qp + (size_t)b * GN * GD;

    for (int half = 0; half < 2; ++half) {
        for (int e = t; e < GN * GN; e += 256) {
            int n = e / GN, m = e % GN;
            float pos[4];
            pos[0] = logf(fmaxf(fabsf((cx[n] - cx[m]) / ww[n]), 1e-3f));
            pos[1] = logf(fmaxf(fabsf((cy[n] - cy[m]) / hh[n]), 1e-3f));
            pos[2] = logf(ww[n] / ww[m]);
            pos[3] = logf(hh[n] / hh[m]);
            float acc[8];
#pragma unroll
            for (int q = 0; q < 8; ++q) acc[q] = bgS[half * 8 + q];
            for (int j = 0; j < 32; ++j) {
                float dm  = __expf(-(float)(j & 7) * 0.8634694f); // 1000^(-(j&7)/8)
                float ang = 100.0f * pos[j >> 3] * dm;
                float s, c;
                __sincosf(ang, &s, &c);
#pragma unroll
                for (int q = 0; q < 8; ++q)
                    acc[q] += s * WgS[(half * 8 + q) * 64 + j] +
                              c * WgS[(half * 8 + q) * 64 + 32 + j];
            }
#pragma unroll
            for (int q = 0; q < 8; ++q)
                g[(q * GN + n) * GN + m] = fmaxf(acc[q], 0.0f);
        }
        __syncthreads();

        for (int idx = t; idx < 8 * GN; idx += 256) {
            int q = idx / GN, m = idx % GN;
            int h = half * 8 + q;
            float accw = 0.0f;
            for (int n = 0; n < GN; ++n) {
                float alpha = baS[m];
                const float* qrow = qb + (size_t)n * GD + h * 64;
                for (int d = 0; d < 64; ++d)
                    alpha += qrow[d] * WaS[d * GN + m];
                accw += alpha * g[(q * GN + n) * GN + m];
            }
            wgacc[h * GN + m] = accw;
        }
        __syncthreads();
    }

    for (int e = t; e < GH * GN; e += 256)
        logw[(size_t)b * (GH * GN) + e] = logf(fmaxf(wgacc[e], 1e-6f));
}

// ===========================================================================
// Attention core: one workgroup per (b,h). Emits f16 context for final GEMM.
// ===========================================================================
__global__ __launch_bounds__(256)
void attn_core(const float* __restrict__ Qp, const float* __restrict__ Kp,
               const float* __restrict__ Vp, const float* __restrict__ logw,
               _Float16* __restrict__ attn) {
    __shared__ float Qs[GN * GDK], Ks[GN * GDK], Vs[GN * GDK];
    __shared__ float S[GN * 40];
    __shared__ float lw[GN];

    const int b = blockIdx.x;
    const int h = blockIdx.y;
    const int t = threadIdx.x;
    const size_t base = (size_t)b * GN * GD + (size_t)h * GDK;

    for (int e = t; e < GN * GDK; e += 256) {
        int n = e >> 6, d = e & 63;
        size_t off = base + (size_t)n * GD + d;
        Qs[e] = Qp[off];
        Ks[e] = Kp[off];
        Vs[e] = Vp[off];
    }
    if (t < GN) lw[t] = logw[((size_t)b * GH + h) * GN + t];
    __syncthreads();

    for (int e = t; e < GN * GN; e += 256) {
        int n = e / GN, m = e % GN;
        float s = 0.0f;
        for (int d = 0; d < GDK; ++d) s += Qs[n * GDK + d] * Ks[m * GDK + d];
        S[n * 40 + m] = s * 0.125f + lw[m];     // 1/sqrt(64)
    }
    __syncthreads();

    if (t < GN) {
        int n = t;
        float mx = -3.4e38f;
        for (int m = 0; m < GN; ++m) mx = fmaxf(mx, S[n * 40 + m]);
        float sum = 0.0f;
        for (int m = 0; m < GN; ++m) {
            float e = __expf(S[n * 40 + m] - mx);
            S[n * 40 + m] = e;
            sum += e;
        }
        float inv = 1.0f / sum;
        for (int m = 0; m < GN; ++m) S[n * 40 + m] *= inv;
    }
    __syncthreads();

    for (int e = t; e < GN * GDK; e += 256) {
        int n = e >> 6, d = e & 63;
        float o = 0.0f;
        for (int m = 0; m < GN; ++m) o += S[n * 40 + m] * Vs[m * GDK + d];
        attn[base + (size_t)n * GD + d] = (_Float16)o;
    }
}

// ===========================================================================
extern "C" void kernel_launch(void* const* d_in, const int* in_sizes, int n_in,
                              void* d_out, int out_size, void* d_ws, size_t ws_size,
                              hipStream_t stream) {
    (void)in_sizes; (void)n_in; (void)out_size; (void)ws_size;

    const float* q   = (const float*)d_in[0];
    const float* k   = (const float*)d_in[1];
    const float* v   = (const float*)d_in[2];
    const float* box = (const float*)d_in[3];
    const float* Wq  = (const float*)d_in[4];
    const float* bq  = (const float*)d_in[5];
    const float* Wk  = (const float*)d_in[6];
    const float* bk  = (const float*)d_in[7];
    const float* Wv  = (const float*)d_in[8];
    const float* bv  = (const float*)d_in[9];
    const float* Wo  = (const float*)d_in[10];
    const float* bo  = (const float*)d_in[11];
    const float* Wg  = (const float*)d_in[12];
    const float* bg  = (const float*)d_in[13];
    const float* Wa  = (const float*)d_in[14];
    const float* ba  = (const float*)d_in[15];
    float* out = (float*)d_out;

    const size_t MD = (size_t)GM * GD;

    char* p = (char*)d_ws;
    float*    Qp  = (float*)p;     p += MD * 4;
    float*    Kp  = (float*)p;     p += MD * 4;
    float*    Vp  = (float*)p;     p += MD * 4;
    _Float16* Ath = (_Float16*)p;  p += MD * 2;
    _Float16* Xh  = (_Float16*)p;  p += MD * 2;
    _Float16* Wt  = (_Float16*)p;  p += (size_t)GD * GD * 2;
    float*    Lw  = (float*)p;     // GB*GH*GN floats

    const dim3 gemm_grid(GM / 128, GD / 128);    // 288 x 8
    const dim3 tr_grid(GD / 32, GD / 32);        // 32 x 32
    const int  cvt_blocks = (int)(MD / 4 / 256); // 36864

    // Q projection
    cvt_t_f16<<<tr_grid, 256, 0, stream>>>(Wq, Wt, GD, GD);
    cvt_f16  <<<cvt_blocks, 256, 0, stream>>>(q, Xh, (int)(MD / 4));
    gemm_f16 <<<gemm_grid, 256, 0, stream>>>(Xh, Wt, bq, Qp, GM, GD, GD);
    // K projection
    cvt_t_f16<<<tr_grid, 256, 0, stream>>>(Wk, Wt, GD, GD);
    cvt_f16  <<<cvt_blocks, 256, 0, stream>>>(k, Xh, (int)(MD / 4));
    gemm_f16 <<<gemm_grid, 256, 0, stream>>>(Xh, Wt, bk, Kp, GM, GD, GD);
    // V projection
    cvt_t_f16<<<tr_grid, 256, 0, stream>>>(Wv, Wt, GD, GD);
    cvt_f16  <<<cvt_blocks, 256, 0, stream>>>(v, Xh, (int)(MD / 4));
    gemm_f16 <<<gemm_grid, 256, 0, stream>>>(Xh, Wt, bv, Vp, GM, GD, GD);

    geom_logw<<<GB, 256, 0, stream>>>(box, Qp, Wg, bg, Wa, ba, Lw);

    attn_core<<<dim3(GB, GH), 256, 0, stream>>>(Qp, Kp, Vp, Lw, Ath);

    // Output projection
    cvt_t_f16<<<tr_grid, 256, 0, stream>>>(Wo, Wt, GD, GD);
    gemm_f16 <<<gemm_grid, 256, 0, stream>>>(Ath, Wt, bo, out, GM, GD, GD);
}
